// SparseLDS_58798102282591
// MI455X (gfx1250) — matrix operand order repaired and checked by reading, not compiled, not run on hardware
//
#include <hip/hip_runtime.h>
#include <hip/hip_bf16.h>

typedef __attribute__((ext_vector_type(16))) _Float16 v16h;
typedef __attribute__((ext_vector_type(8)))  float    v8f;

#define B_   8
#define T_   8192
#define N_   512
#define M_   512
#define CHUNKS 16
#define CLEN 512   // T_/CHUNKS
#define KTOT 1024  // 2*N_ (re | im)

// ---------------------------------------------------------------------------
// Kernel 1: chunked diagonal-complex scan.
// grid = (CHUNKS, B_), block = 512 (one thread per state n).
// x chunk staged in LDS (broadcast reads), local recurrence in registers,
// chunk result rescaled by lam^(T-end) via repeated squaring (exact).
// ---------------------------------------------------------------------------
__global__ __launch_bounds__(512)
void SparseLDS_scan_kernel(const float* __restrict__ x,
                           const float* __restrict__ lre,
                           const float* __restrict__ lim,
                           float* __restrict__ pre,
                           float* __restrict__ pim) {
    __shared__ float xs[CLEN];
    const int c   = blockIdx.x;   // chunk index
    const int bb  = blockIdx.y;   // batch
    const int tid = threadIdx.x;  // state n

    xs[tid] = x[bb * T_ + c * CLEN + tid];
    __syncthreads();

    const float lr = lre[tid], li = lim[tid];
    const float er = __expf(lr);
    const float lamr = er * cosf(li);
    const float lami = er * sinf(li);

    float sr = 0.f, si = 0.f;
    #pragma unroll 8
    for (int t = 0; t < CLEN; ++t) {
        const float xv = xs[t];
        const float nr = fmaf(lamr, sr, fmaf(-lami, si, xv));
        const float ni = fmaf(lamr, si, lami * sr);
        sr = nr; si = ni;
    }

    // scale = (lam^512)^(CHUNKS-1-c)
    float qr = lamr, qi = lami;
    for (int i = 0; i < 9; ++i) {
        const float tr = qr * qr - qi * qi;
        const float ti = 2.f * qr * qi;
        qr = tr; qi = ti;
    }
    float scr = 1.f, sci = 0.f;
    for (int i = 0; i < CHUNKS - 1 - c; ++i) {
        const float tr = scr * qr - sci * qi;
        const float ti = scr * qi + sci * qr;
        scr = tr; sci = ti;
    }

    const int o = (c * B_ + bb) * N_ + tid;
    pre[o] = scr * sr - sci * si;
    pim[o] = scr * si + sci * sr;
}

// ---------------------------------------------------------------------------
// Kernel 2: reduce chunk contributions -> s_T [B_ x N_] (re, im).
// ---------------------------------------------------------------------------
__global__ __launch_bounds__(256)
void SparseLDS_reduce_kernel(const float* __restrict__ pre,
                             const float* __restrict__ pim,
                             float* __restrict__ str,
                             float* __restrict__ sti) {
    const int idx = blockIdx.x * blockDim.x + threadIdx.x;  // bb*N_ + nn
    if (idx < B_ * N_) {
        float sr = 0.f, si = 0.f;
        #pragma unroll
        for (int cc = 0; cc < CHUNKS; ++cc) {
            sr += pre[cc * (B_ * N_) + idx];
            si += pim[cc * (B_ * N_) + idx];
        }
        str[idx] = sr;
        sti[idx] = si;
    }
}

// ---------------------------------------------------------------------------
// Kernel 3: pre-pack A fragments (f16) in exact per-lane WMMA layout.
// A' = [sT_re | sT_im], 16 rows (8 valid) x 1024 cols.
// afrag[(kk*32 + lane)*16 + h] with k = 16*(lane/16) + h (contiguous order, as B uses).
// Branches are fine here: 1024 threads, runs once, output is 32 KB.
// ---------------------------------------------------------------------------
__global__ __launch_bounds__(256)
void SparseLDS_packA_kernel(const float* __restrict__ str,
                            const float* __restrict__ sti,
                            _Float16* __restrict__ afrag) {
    const int idx = blockIdx.x * blockDim.x + threadIdx.x;  // 0..1023
    if (idx >= 32 * 32) return;
    const int kk   = idx >> 5;         // k-step 0..31
    const int lane = idx & 31;         // lane 0..31
    const int Mrow = lane & 15;
    const int kbA  = (lane & 16) ? 16 : 0;

    v16h v;
    #pragma unroll
    for (int h = 0; h < 16; ++h) {
        const int j  = h >> 1;
        const int lo = h & 1;
        const int kl = kbA + 2 * j + lo;
        const int Ka = kk * 32 + kl;   // 0..1023
        float av = 0.f;
        if (Mrow < B_)
            av = (Ka < N_) ? str[Mrow * N_ + Ka] : sti[Mrow * N_ + (Ka - N_)];
        v[h] = (_Float16)av;
    }
    *(v16h*)(afrag + idx * 16) = v;    // 32B aligned vector store
}

// ---------------------------------------------------------------------------
// Kernel 4: y = Re(sT @ C) + x_last @ D + Do via v_wmma_f32_16x16x32_f16.
// One wave32 per 16-column tile. A fragments: 2x b128 loads from afrag.
// B fragments: k-loop split so steps 0..15 touch only C_re, 16..31 only -C_im
// (no per-element conditionals -> no exec-mask branching in the hot loop).
// ---------------------------------------------------------------------------
__global__ __launch_bounds__(32)
void SparseLDS_wmma_kernel(const _Float16* __restrict__ afrag,
                           const float* __restrict__ Cre,
                           const float* __restrict__ Cim,
                           const float* __restrict__ x,
                           const float* __restrict__ Dm,
                           const float* __restrict__ Do,
                           float* __restrict__ y) {
    const int nTile = blockIdx.x;            // 0..31
    const int lane  = threadIdx.x;           // 0..31 (wave32, EXEC all ones)
    const int kbB   = (lane & 16) ? 16 : 0;  // B K-group base (contiguous k order, as afrag)
    const int col   = nTile * 16 + (lane & 15);

    // Per-lane column pointers (row stride M_ floats).
    const float* __restrict__ bre = Cre + col;
    const float* __restrict__ bim = Cim + col;

    v8f acc = {};

    // k-steps 0..15: K in [0,512) -> +C_re
    for (int kk = 0; kk < 16; ++kk) {
        const int rowbase = kk * 32 + kbB;
        v16h b;
        #pragma unroll
        for (int h = 0; h < 16; ++h)
            b[h] = (_Float16)bre[(rowbase + h) * M_];
        const v16h a = *(const v16h*)(afrag + (kk * 32 + lane) * 16);
        acc = __builtin_amdgcn_wmma_f32_16x16x32_f16(
            false, a, false, b, (short)0, acc, false, false);
    }
    // k-steps 16..31: K in [512,1024) -> -C_im
    for (int kk = 16; kk < 32; ++kk) {
        const int rowbase = (kk - 16) * 32 + kbB;
        v16h b;
        #pragma unroll
        for (int h = 0; h < 16; ++h)
            b[h] = (_Float16)(-bim[(rowbase + h) * M_]);
        const v16h a = *(const v16h*)(afrag + (kk * 32 + lane) * 16);
        acc = __builtin_amdgcn_wmma_f32_16x16x32_f16(
            false, a, false, b, (short)0, acc, false, false);
    }

    // D-matrix layout: lanes 0-15 hold rows M = v (VGPR index 0..7) — exactly
    // the 8 valid batch rows. Lanes 16-31 hold padding rows 8..15.
    if (lane < 16) {
        const float dcol = Dm[col];
        const float ocol = Do[col];
        #pragma unroll
        for (int v = 0; v < 8; ++v) {
            const float xl = x[v * T_ + (T_ - 1)];   // x[:, -1, 0]
            y[v * M_ + col] = acc[v] + xl * dcol + ocol;
        }
    }
}

extern "C" void kernel_launch(void* const* d_in, const int* in_sizes, int n_in,
                              void* d_out, int out_size, void* d_ws, size_t ws_size,
                              hipStream_t stream) {
    const float* x    = (const float*)d_in[0];  // [8, 8192, 1]
    const float* lre  = (const float*)d_in[1];  // [512]
    const float* lim  = (const float*)d_in[2];  // [512]
    const float* Cre  = (const float*)d_in[3];  // [512, 512]
    const float* Cim  = (const float*)d_in[4];  // [512, 512]
    const float* Dm   = (const float*)d_in[5];  // [1, 512]
    const float* Do   = (const float*)d_in[6];  // [512]
    float* y = (float*)d_out;                   // [8, 512]

    // Workspace carve-out: partials (512KB) + sT (32KB) + packed A (32KB).
    float* w   = (float*)d_ws;
    float* pre = w;                                    // 65536 floats
    float* pim = w + CHUNKS * B_ * N_;                 // 65536 floats
    float* str = w + 2 * CHUNKS * B_ * N_;             // 4096 floats
    float* sti = str + B_ * N_;                        // 4096 floats
    _Float16* afrag = (_Float16*)(sti + B_ * N_);      // 16384 halves (32KB)

    SparseLDS_scan_kernel<<<dim3(CHUNKS, B_), 512, 0, stream>>>(x, lre, lim, pre, pim);
    SparseLDS_reduce_kernel<<<(B_ * N_ + 255) / 256, 256, 0, stream>>>(pre, pim, str, sti);
    SparseLDS_packA_kernel<<<(32 * 32 + 255) / 256, 256, 0, stream>>>(str, sti, afrag);
    SparseLDS_wmma_kernel<<<M_ / 16, 32, 0, stream>>>(afrag, Cre, Cim, x, Dm, Do, y);
}
